// TopkSparseAutoencoder_49667001811571
// MI455X (gfx1250) — matrix operand order repaired and checked by reading, not compiled, not run on hardware
//
#include <hip/hip_runtime.h>
#include <hip/hip_bf16.h>

// ---------------- problem dims ----------------
#define BATCH   8192
#define FEAT    1024      // K-dim of encoder GEMM, also decode output width
#define PAGES   16384
#define TOPK    32

#define ROWS_PER_WG  64
#define PAGE_TILE    32                    // pages per chunk (2 waves x 16)
#define NCHUNK       (PAGES / PAGE_TILE)   // 512
#define THREADS      256                   // 8 waves (wave32)

// ---------------- workspace layout (bytes) ----------------
#define XH_OFF   (0)                                    // f16 [BATCH][FEAT]      16 MiB
#define WH_OFF   (XH_OFF + (size_t)BATCH*FEAT*2)        // f16 [PAGES][FEAT]      32 MiB
#define DWT_OFF  (WH_OFF + (size_t)PAGES*FEAT*2)        // f16 [PAGES][FEAT] (dec_w^T) 32 MiB
#define TKV_OFF  (DWT_OFF + (size_t)PAGES*FEAT*2)       // f32 [BATCH][TOPK]       1 MiB
#define TKI_OFF  (TKV_OFF + (size_t)BATCH*TOPK*4)       // i32 [BATCH][TOPK]       1 MiB

typedef __attribute__((ext_vector_type(4)))  _Float16 v4h;
typedef __attribute__((ext_vector_type(8)))  _Float16 v8h;
typedef __attribute__((ext_vector_type(16))) _Float16 v16h;
typedef __attribute__((ext_vector_type(8)))  float    v8f;
typedef __attribute__((ext_vector_type(4)))  int      v4i;

typedef __attribute__((address_space(1))) v4i g_v4i;
typedef __attribute__((address_space(3))) v4i l_v4i;

// ---------------- CDNA5 async global->LDS copy (ASYNCcnt path) ----------------
#if __has_builtin(__builtin_amdgcn_global_load_async_to_lds_b128)
#define HAVE_ASYNC_LDS 1
#else
#define HAVE_ASYNC_LDS 0
#endif

__device__ __forceinline__ void copy16_async(void* lds, const void* g) {
#if HAVE_ASYNC_LDS
    __builtin_amdgcn_global_load_async_to_lds_b128(
        (g_v4i*)const_cast<void*>(g),
        (l_v4i*)lds,
        0, 0);
#else
    *(v8h*)lds = *(const v8h*)g;   // sync fallback: global_load_b128 + ds_store_b128
#endif
}

__device__ __forceinline__ void wait_async_all() {
#if HAVE_ASYNC_LDS
#if __has_builtin(__builtin_amdgcn_s_wait_asynccnt)
    __builtin_amdgcn_s_wait_asynccnt(0);
#else
    asm volatile("s_wait_asynccnt 0x0" ::: "memory");
#endif
#endif
}

// ---------------- prep: xh = f16(x - bias) ----------------
__global__ void prep_x_kernel(const float* __restrict__ x,
                              const float* __restrict__ bias,
                              _Float16* __restrict__ xh) {
    int i = blockIdx.x * blockDim.x + threadIdx.x;   // i < BATCH*FEAT
    int f = i & (FEAT - 1);
    xh[i] = (_Float16)(x[i] - bias[f]);
}

// ---------------- prep: wh = f16(enc_w) ----------------
__global__ void prep_w_kernel(const float* __restrict__ w,
                              _Float16* __restrict__ wh) {
    int i = blockIdx.x * blockDim.x + threadIdx.x;   // i < PAGES*FEAT
    wh[i] = (_Float16)w[i];
}

// ---------------- prep: dwt[p][f] = f16(dec_w[f][p]) ----------------
__global__ void prep_dec_kernel(const float* __restrict__ dw,
                                _Float16* __restrict__ dwt) {
    int i = blockIdx.x * blockDim.x + threadIdx.x;   // output-linear: i < PAGES*FEAT
    int p = i >> 10;             // FEAT == 1024
    int f = i & (FEAT - 1);
    dwt[i] = (_Float16)dw[(size_t)f * PAGES + p];
}

// ---------------- fused encode GEMM + relu + top-32 ----------------
// grid = BATCH/64 workgroups, block = 256 threads (8 waves, wave32).
// Waves form a 4 (row tiles) x 2 (page tiles) grid: workgroup owns 64 rows,
// each chunk covers 32 pages whose B tile is double-buffered in LDS and
// filled via async global->LDS loads overlapped with the previous chunk's
// WMMA compute. One thread per row maintains its top-32 in LDS.
__global__ void encode_topk_kernel(const _Float16* __restrict__ xh,
                                   const _Float16* __restrict__ wh,
                                   const float* __restrict__ enc_b,
                                   float* __restrict__ tkv_g,
                                   int*   __restrict__ tki_g) {
    extern __shared__ char smem[];
    _Float16* Ash  = (_Float16*)smem;                                // 64*1024 f16 = 128 KiB
    _Float16* Bsh0 = (_Float16*)(smem + (size_t)ROWS_PER_WG*FEAT*2); // 32*1024 f16 =  64 KiB
    _Float16* Bsh1 = Bsh0 + (size_t)PAGE_TILE*FEAT;                  //               64 KiB
    float* cand = (float*)((char*)Bsh1 + (size_t)PAGE_TILE*FEAT*2);  // 64*32 f32 = 8 KiB
    float* tkv  = (float*)((char*)cand + ROWS_PER_WG*PAGE_TILE*4);   // 64*32 f32 = 8 KiB
    int*   tki  = (int*)((char*)tkv + ROWS_PER_WG*TOPK*4);           // 64*32 i32 = 8 KiB

    const int t    = threadIdx.x;
    const int wave = t >> 5;            // 0..7
    const int lane = t & 31;
    const int hi   = (lane >= 16) ? 1 : 0;
    const int ln16 = lane & 15;
    const int rowTile  = wave & 3;      // 0..3 -> rows [rowTile*16, +16)
    const int pageTile = wave >> 2;     // 0..1 -> pages [pageTile*16, +16)
    const int row0 = blockIdx.x * ROWS_PER_WG;

    // ---- issue async load of A tile: 64 rows x 1024 f16 (128 KiB) ----
    {
        const char* srcA = (const char*)(xh + (size_t)row0 * FEAT);
        #pragma unroll
        for (int j = 0; j < 32; ++j) {
            size_t off = (size_t)(t + j * THREADS) * 16;
            copy16_async((char*)Ash + off, srcA + off);
        }
    }
    // ---- issue async load of B chunk 0 into buffer 0 ----
    {
        const char* srcB = (const char*)wh;
        #pragma unroll
        for (int j = 0; j < 16; ++j) {
            size_t off = (size_t)(t + j * THREADS) * 16;
            copy16_async((char*)Bsh0 + off, srcB + off);
        }
    }

    // ---- init topk state ----
    for (int j = t; j < ROWS_PER_WG * TOPK; j += THREADS) { tkv[j] = 0.0f; tki[j] = 0; }
    float minv = 0.0f;   // running min of this thread's row (threads t<64)
    int   minpos = 0;

    const int arow = rowTile * 16 + ln16;
    const _Float16* Arow = Ash + (size_t)arow * FEAT;
    const int sel8 = hi ? 8 : 0;
    const int bcol = pageTile * 16 + ln16;

    for (int ci = 0; ci < NCHUNK; ++ci) {
        const int pc = ci * PAGE_TILE;
        _Float16* Bcur = (ci & 1) ? Bsh1 : Bsh0;
        _Float16* Bnxt = (ci & 1) ? Bsh0 : Bsh1;

        wait_async_all();      // this wave's portions of A (first iter) + Bcur landed
        __syncthreads();       // all waves' portions landed; Bnxt no longer being read

        // ---- issue async load of next B chunk into the other buffer ----
        if (ci + 1 < NCHUNK) {
            const char* srcB = (const char*)(wh + (size_t)(pc + PAGE_TILE) * FEAT);
            #pragma unroll
            for (int j = 0; j < 16; ++j) {
                size_t off = (size_t)(t + j * THREADS) * 16;
                copy16_async((char*)Bnxt + off, srcB + off);
            }
        }

        // ---- GEMM: 16 rows x 16 pages (this wave), K = 1024 ----
        v8f acc = {};
        const _Float16* Bcol = Bcur + (size_t)bcol * FEAT;
        #pragma unroll 4
        for (int kk = 0; kk < FEAT; kk += 32) {
            // A 16x32 f16 layout: lane<16 holds K[0..15], lane>=16 holds K[16..31] (contiguous)
            v16h a = *(const v16h*)(Arow + kk + (hi ? 16 : 0));
            // B 32x16 f16 layout: lane n holds K[0..15], lane n+16 holds K[16..31] (contiguous)
            v16h b = *(const v16h*)(Bcol + kk + (hi ? 16 : 0));
            acc = __builtin_amdgcn_wmma_f32_16x16x32_f16(
                      false, a, false, b, (short)0, acc, false, false);
        }

        // ---- bias + relu, stage candidates ----
        {
            float eb = enc_b[pc + bcol];
            #pragma unroll
            for (int j = 0; j < 8; ++j) {
                float v = acc[j] + eb;
                v = v > 0.0f ? v : 0.0f;
                int lrow = rowTile * 16 + (hi ? j + 8 : j);
                cand[lrow * PAGE_TILE + bcol] = v;
            }
        }
        __syncthreads();

        // ---- top-32 merge: thread t owns row t (t < 64) ----
        if (t < ROWS_PER_WG) {
            #pragma unroll
            for (int c = 0; c < PAGE_TILE; ++c) {
                float v = cand[t * PAGE_TILE + c];
                if (v > minv) {
                    tkv[t * TOPK + minpos] = v;
                    tki[t * TOPK + minpos] = pc + c;
                    float m = tkv[t * TOPK];
                    int mp = 0;
                    #pragma unroll
                    for (int j = 1; j < TOPK; ++j) {
                        float tvj = tkv[t * TOPK + j];
                        if (tvj < m) { m = tvj; mp = j; }
                    }
                    minv = m; minpos = mp;
                }
            }
        }
    }
    __syncthreads();

    // ---- write out per-row top-32 ----
    if (t < ROWS_PER_WG) {
        int row = row0 + t;
        #pragma unroll
        for (int j = 0; j < TOPK; ++j) {
            tkv_g[(size_t)row * TOPK + j] = tkv[t * TOPK + j];
            tki_g[(size_t)row * TOPK + j] = tki[t * TOPK + j];
        }
    }
}

// ---------------- sparse decode: out[b,:] = bias + sum_k v_k * dec_wT[p_k,:] ----------------
__global__ void decode_kernel(const float* __restrict__ tkv_g,
                              const int*   __restrict__ tki_g,
                              const _Float16* __restrict__ dwt,
                              const float* __restrict__ bias,
                              float* __restrict__ out) {
    const int b = blockIdx.x;
    const int t = threadIdx.x;          // 256 threads, 4 features each
    __shared__ float sv[TOPK];
    __shared__ int   si[TOPK];
    if (t < TOPK) {
        sv[t] = tkv_g[(size_t)b * TOPK + t];
        si[t] = tki_g[(size_t)b * TOPK + t];
    }
    __syncthreads();

    const int f = t * 4;
    float a0 = 0.f, a1 = 0.f, a2 = 0.f, a3 = 0.f;
    #pragma unroll
    for (int k = 0; k < TOPK; ++k) {
        float v = sv[k];
        int   p = si[k];
        v4h w = *(const v4h*)(dwt + (size_t)p * FEAT + f);
        a0 += v * (float)w[0];
        a1 += v * (float)w[1];
        a2 += v * (float)w[2];
        a3 += v * (float)w[3];
    }
    float4 bb = *(const float4*)(bias + f);
    float4 r;
    r.x = a0 + bb.x; r.y = a1 + bb.y; r.z = a2 + bb.z; r.w = a3 + bb.w;
    *(float4*)(out + (size_t)b * FEAT + f) = r;
}

extern "C" void kernel_launch(void* const* d_in, const int* in_sizes, int n_in,
                              void* d_out, int out_size, void* d_ws, size_t ws_size,
                              hipStream_t stream) {
    const float* x     = (const float*)d_in[0];   // [BATCH, FEAT]
    const float* enc_w = (const float*)d_in[1];   // [PAGES, FEAT]
    const float* enc_b = (const float*)d_in[2];   // [PAGES]
    const float* dec_w = (const float*)d_in[3];   // [FEAT, PAGES]
    const float* bias  = (const float*)d_in[4];   // [FEAT]
    float* out = (float*)d_out;

    char* ws = (char*)d_ws;
    _Float16* xh  = (_Float16*)(ws + XH_OFF);
    _Float16* wh  = (_Float16*)(ws + WH_OFF);
    _Float16* dwt = (_Float16*)(ws + DWT_OFF);
    float*    tkv = (float*)(ws + TKV_OFF);
    int*      tki = (int*)(ws + TKI_OFF);

    // prep / precision conversion
    prep_x_kernel<<<(BATCH * FEAT) / 256, 256, 0, stream>>>(x, bias, xh);
    prep_w_kernel<<<(PAGES * FEAT) / 256, 256, 0, stream>>>(enc_w, wh);
    prep_dec_kernel<<<(PAGES * FEAT) / 256, 256, 0, stream>>>(dec_w, dwt);

    // fused encoder GEMM (WMMA f16) + relu + top-32
    size_t smem = (size_t)ROWS_PER_WG * FEAT * 2        // A tile        128 KiB
                + (size_t)2 * PAGE_TILE * FEAT * 2      // B double buf  128 KiB
                + (size_t)ROWS_PER_WG * PAGE_TILE * 4   // candidates      8 KiB
                + (size_t)ROWS_PER_WG * TOPK * 4        // topk vals       8 KiB
                + (size_t)ROWS_PER_WG * TOPK * 4;       // topk idx        8 KiB
    encode_topk_kernel<<<BATCH / ROWS_PER_WG, THREADS, smem, stream>>>(xh, wh, enc_b, tkv, tki);

    // sparse decode
    decode_kernel<<<BATCH, 256, 0, stream>>>(tkv, tki, dwt, bias, out);
}
